// SimpleBioInspiredModel_49718541418734
// MI455X (gfx1250) — matrix operand order repaired and checked by reading, not compiled, not run on hardware
//
#include <hip/hip_runtime.h>
#include <hip/hip_bf16.h>

// ---------------- model constants ----------------
#define BATCH   2048
#define FEAT    2048
#define HARM    32
#define DIN     2112          // FEAT + 2*HARM
#define NEXP    8
#define EDIM    32
#define HID     1024
#define VOCAB   50257
#define NPAD    50304         // VOCAB padded to multiple of 128
#define TWO_PI_D0 43.9822971502571f   // 2*pi*7

#define KC        32          // K-chunk per pipeline stage
#define LDS_PITCH 40          // 32 bf16 (64B) + 16B pad -> 80B pitch, conflict-free b128 reads

typedef __attribute__((ext_vector_type(16))) __bf16 bf16x16;
typedef __attribute__((ext_vector_type(8)))  __bf16 bf16x8;
typedef __attribute__((ext_vector_type(8)))  float  f32x8;
typedef __attribute__((ext_vector_type(4)))  unsigned int u32x4;
typedef __attribute__((ext_vector_type(8)))  int i32x8;
typedef __attribute__((ext_vector_type(4)))  int i32x4;

#if defined(__gfx1250__) && __has_builtin(__builtin_amdgcn_tensor_load_to_lds) && \
    __has_builtin(__builtin_amdgcn_s_wait_tensorcnt)
#define HAVE_TDM 1
#else
#define HAVE_TDM 0
#endif

// =====================================================================
// Kernel 1: per-row mean -> phasor features [B,64]
// =====================================================================
__global__ __launch_bounds__(256) void k_mean_phasor(const float* __restrict__ x,
                                                     float* __restrict__ phasor) {
    __shared__ float red[256];
    const int b = blockIdx.x, tid = threadIdx.x;
    float s = 0.f;
    for (int i = tid; i < FEAT; i += 256) s += x[(size_t)b * FEAT + i];
    red[tid] = s; __syncthreads();
    for (int off = 128; off; off >>= 1) {
        if (tid < off) red[tid] += red[tid + off];
        __syncthreads();
    }
    const float mean = red[0] * (1.0f / (float)FEAT);
    if (tid < 2 * HARM) {
        const int k = (tid & (HARM - 1)) + 1;
        const float ph = TWO_PI_D0 * mean * (float)k;
        phasor[b * 2 * HARM + tid] = (tid < HARM) ? cosf(ph) : sinf(ph);
    }
}

// =====================================================================
// Kernel 2: gate softmax + expert einsum + mixture -> mixed [B,32]
// R=4 rows per block; enhanced rows cached in LDS so We is read 512x total.
// =====================================================================
__global__ __launch_bounds__(256) void k_gate_mix(const float* __restrict__ x,
                                                  const float* __restrict__ phasor,
                                                  const float* __restrict__ Wg,
                                                  const float* __restrict__ bg,
                                                  const float* __restrict__ We,
                                                  const float* __restrict__ be,
                                                  float* __restrict__ mixed) {
    __shared__ float ls_enh[4][DIN];
    __shared__ float ls_eo[4][NEXP][EDIM];
    __shared__ float ls_gl[4][NEXP];
    __shared__ float ls_gate[4][NEXP];
    const int tid = threadIdx.x;
    const int b0 = blockIdx.x * 4;

    for (int r = 0; r < 4; ++r) {
        const int b = b0 + r;
        for (int i = tid; i < DIN; i += 256)
            ls_enh[r][i] = (i < FEAT) ? x[(size_t)b * FEAT + i]
                                      : phasor[b * 2 * HARM + (i - FEAT)];
    }
    __syncthreads();

    { // expert outputs: thread -> (e = tid/32, h = tid%32), 4 rows each
        const int e = tid >> 5, h = tid & 31;
        float a0 = 0.f, a1 = 0.f, a2 = 0.f, a3 = 0.f;
        const float* wp = We + ((size_t)e * DIN) * EDIM + h;
        for (int d = 0; d < DIN; ++d) {
            const float w = wp[(size_t)d * EDIM];
            a0 += w * ls_enh[0][d]; a1 += w * ls_enh[1][d];
            a2 += w * ls_enh[2][d]; a3 += w * ls_enh[3][d];
        }
        const float bb = be[e * EDIM + h];
        ls_eo[0][e][h] = a0 + bb; ls_eo[1][e][h] = a1 + bb;
        ls_eo[2][e][h] = a2 + bb; ls_eo[3][e][h] = a3 + bb;
    }
    if (tid < 32) { // gate logits: (r = tid/8, e = tid%8)
        const int r = tid >> 3, e = tid & 7;
        float g = 0.f;
        for (int d = 0; d < DIN; ++d) g += Wg[d * NEXP + e] * ls_enh[r][d];
        ls_gl[r][e] = g + bg[e];
    }
    __syncthreads();
    if (tid < 4) { // softmax over 8 experts
        const int r = tid;
        float m = ls_gl[r][0];
        for (int e = 1; e < NEXP; ++e) m = fmaxf(m, ls_gl[r][e]);
        float sum = 0.f;
        for (int e = 0; e < NEXP; ++e) { const float t = expf(ls_gl[r][e] - m); ls_gate[r][e] = t; sum += t; }
        const float inv = 1.0f / sum;
        for (int e = 0; e < NEXP; ++e) ls_gate[r][e] *= inv;
    }
    __syncthreads();
    if (tid < 128) { // mixed[r][h] = sum_e gate*eo
        const int r = tid >> 5, h = tid & 31;
        float m = 0.f;
        for (int e = 0; e < NEXP; ++e) m += ls_gate[r][e] * ls_eo[r][e][h];
        mixed[(b0 + r) * EDIM + h] = m;
    }
}

// =====================================================================
// Kernel 3: ctx = tanh(mixed @ Wo + bo)   [B,1024]
// =====================================================================
__global__ __launch_bounds__(256) void k_ctx(const float* __restrict__ mixed,
                                             const float* __restrict__ Wo,
                                             const float* __restrict__ bo,
                                             float* __restrict__ ctx) {
    __shared__ float ls_m[EDIM];
    const int b = blockIdx.x, tid = threadIdx.x;
    if (tid < EDIM) ls_m[tid] = mixed[b * EDIM + tid];
    __syncthreads();
    for (int c = tid; c < HID; c += 256) {
        float acc = bo[c];
        for (int k = 0; k < EDIM; ++k) acc += ls_m[k] * Wo[k * HID + c];
        ctx[(size_t)b * HID + c] = tanhf(acc);
    }
}

// =====================================================================
// Kernel 4 (single block): top-20 of ctx[0,:] (argsort-tail semantics)
// + 20-step LIF / 5-winner-take-all spiking sim -> gains[1024]
// =====================================================================
__global__ __launch_bounds__(256) void k_spike(const float* __restrict__ ctx,
                                               float* __restrict__ gains) {
    __shared__ float lsv[HID];
    __shared__ float rv[256];
    __shared__ int   ri[256];
    __shared__ int   tok_desc[20];
    const int tid = threadIdx.x;
    for (int i = tid; i < HID; i += 256) { lsv[i] = ctx[i]; gains[i] = 1.0f; }
    __syncthreads();

    // 20 rounds of parallel argmax; tie -> LARGER index (argsort-ascending tail)
    for (int round = 0; round < 20; ++round) {
        float bv = -1e30f; int bi = -1;
        for (int c = tid; c < HID; c += 256)
            if (lsv[c] >= bv) { bv = lsv[c]; bi = c; }
        rv[tid] = bv; ri[tid] = bi; __syncthreads();
        for (int off = 128; off; off >>= 1) {
            if (tid < off) {
                const float ov = rv[tid + off]; const int oi = ri[tid + off];
                if (ov > rv[tid] || (ov == rv[tid] && oi > ri[tid])) { rv[tid] = ov; ri[tid] = oi; }
            }
            __syncthreads();
        }
        if (tid == 0) { tok_desc[round] = ri[0]; lsv[ri[0]] = -1e30f; }
        __syncthreads();
    }

    if (tid == 0) {
        // token_seq = reverse(desc) : ascending value order, ends at the max
        int   tk[20]; float v[20]; float cnt[20]; bool picked[20];
        for (int i = 0; i < 20; ++i) { tk[i] = tok_desc[19 - i]; v[i] = 0.f; cnt[i] = 0.f; }
        for (int step = 0; step < 20; ++step) {
            for (int j = 0; j < 20; ++j) v[j] *= 0.7f;   // DECAY
            v[step] += 1.0f;                              // distinct tokens: slot==step
            for (int j = 0; j < 20; ++j) picked[j] = false;
            for (int p = 0; p < 5; ++p) {                 // K_WINNERS; ties -> lower vocab idx
                int   bj = -1; float bv = -1.0f;
                for (int j = 0; j < 20; ++j) {
                    if (picked[j]) continue;
                    if (v[j] > bv || (v[j] == bv && bj >= 0 && tk[j] < tk[bj])) { bv = v[j]; bj = j; }
                }
                if (bj < 0) break;
                picked[bj] = true;
                if (v[bj] >= 1.0f) { cnt[bj] += 1.0f; v[bj] = 0.0f; }  // THETA spike + reset
            }
        }
        for (int j = 0; j < 20; ++j)
            if (cnt[j] > 0.f) gains[tk[j]] += cnt[j];
    }
}

// =====================================================================
// Kernel 5: attended = ctx * gains -> bf16 A matrix [2048,1024]
// =====================================================================
__global__ __launch_bounds__(256) void k_attend_bf16(const float* __restrict__ ctx,
                                                     const float* __restrict__ gains,
                                                     __bf16* __restrict__ Abf) {
    const size_t idx = (size_t)blockIdx.x * 256 + threadIdx.x;
    const int c = (int)(idx & (HID - 1));
    Abf[idx] = (__bf16)(ctx[idx] * gains[c]);
}

// =====================================================================
// Kernel 6: Wout[1024,50257] fp32 -> Bt[NPAD,1024] bf16 (transposed, padded)
// =====================================================================
__global__ __launch_bounds__(256) void k_conv_wout(const float* __restrict__ Wout,
                                                   __bf16* __restrict__ Bt) {
    __shared__ __bf16 tile[64][72];
    const int tid = threadIdx.x;
    const int n0 = blockIdx.x * 64;   // vocab tile
    const int k0 = blockIdx.y * 64;   // hidden tile
    for (int i = 0; i < 16; ++i) {
        const int lin = tid + i * 256;         // 0..4095
        const int k = lin >> 6, n = lin & 63;  // read coalesced along n
        float v = 0.f;
        if (n0 + n < VOCAB) v = Wout[(size_t)(k0 + k) * VOCAB + (n0 + n)];
        tile[n][k] = (__bf16)v;
    }
    __syncthreads();
    for (int i = 0; i < 16; ++i) {
        const int lin = tid + i * 256;
        const int n = lin >> 6, k = lin & 63;  // write coalesced along k
        Bt[(size_t)(n0 + n) * HID + (k0 + k)] = tile[n][k];
    }
}

// =====================================================================
// TDM: issue a 2-D tile load (128 rows x 32 bf16) into LDS with padding
// producing an 80B row pitch.  D# per CDNA5 ISA §8.3/8.4.
// 6-arg builtin form: (g0, g1, g2, g3, g4, cpol); groups 2-4 zero (2-D).
// Assumes the single __shared__ pool allocates at LDS offset 0.
// =====================================================================
#if HAVE_TDM
static __device__ __forceinline__ void tdm_load_tile(unsigned lds_byte_off,
                                                     const __bf16* gtile,
                                                     unsigned tensor_rows,
                                                     unsigned row_stride_elems) {
    const unsigned long long ga = (unsigned long long)(const void*)gtile;
    u32x4 g0;
    g0[0] = 1u;                                   // count=1 (valid user descriptor)
    g0[1] = lds_byte_off;                         // lds_addr
    g0[2] = (unsigned)ga;                         // global_addr[31:0]
    g0[3] = (unsigned)(ga >> 32) | (2u << 30);    // global_addr[56:32] | type=2 (image)
    i32x8 g1;
    g1[0] = (1 << 16)        // data_size = 2 bytes
          | (1 << 20)        // pad_enable
          | (3 << 22)        // pad_interval = 16 DWORDs (64B row)
          | (3 << 25);       // pad_amount   = 4 DWORDs (16B) -> 80B LDS pitch
    g1[1] = (int)((row_stride_elems & 0xFFFFu) << 16);                  // tensor_dim0[15:0]
    g1[2] = (int)((row_stride_elems >> 16) | ((tensor_rows & 0xFFFFu) << 16)); // dim0[31:16]|dim1[15:0]
    g1[3] = (int)((tensor_rows >> 16) | ((unsigned)KC << 16));          // dim1[31:16]|tile_dim0
    g1[4] = 128;                                                        // tile_dim1=128, tile_dim2=0
    g1[5] = (int)row_stride_elems;                                      // tensor_dim0_stride[31:0]
    g1[6] = 0;
    g1[7] = 0;
    const i32x4 z4 = {0, 0, 0, 0};
    const i32x8 z8 = {0, 0, 0, 0, 0, 0, 0, 0};
    __builtin_amdgcn_tensor_load_to_lds(g0, g1, z4, z4, z8, 0);
}
#endif

// =====================================================================
// Kernel 7: out = A(bf16)[2048,1024] x Bt(bf16)[NPAD,1024]^T + bout
// 128x128 block tile, 8 wave32s (4x2), 2x4 WMMA accs per wave.
// TDM double-buffered K pipeline: wave0 DMAs chunk i+1 while all waves
// run WMMA on chunk i; s_wait_tensorcnt + barrier per stage.
// =====================================================================
static __device__ __forceinline__ bf16x16 frag_cat(bf16x8 lo, bf16x8 hi) {
    return __builtin_shufflevector(lo, hi, 0,1,2,3,4,5,6,7,8,9,10,11,12,13,14,15);
}

__global__ __launch_bounds__(256) void k_gemm_wmma(const __bf16* __restrict__ A,
                                                   const __bf16* __restrict__ Bt,
                                                   const float* __restrict__ bout,
                                                   float* __restrict__ out) {
    // pool[buf][0]=A tile, pool[buf][1]=B tile; 40,960 bytes total
    __shared__ __align__(16) __bf16 pool[2][2][128][LDS_PITCH];
    const int tid  = threadIdx.x;
    const int n0   = blockIdx.x * 128;
    const int m0   = blockIdx.y * 128;
    const int wave = tid >> 5, lane = tid & 31;
    const int wm   = wave & 3;       // 4 waves along M (32 rows each)
    const int wn   = wave >> 2;      // 2 waves along N (64 cols each)
    const int hh   = lane >> 4;      // lane half selector
    const int l15  = lane & 15;
    const unsigned tile_bytes = 128u * LDS_PITCH * 2u;   // 10240

    // TDM writes LDS behind the optimizer's back: escape the pool address and
    // clobber memory so ds_load fragments are neither folded nor hoisted.
#define TDM_LDS_FENCE() asm volatile("" :: "v"((void*)&pool[0][0][0][0]) : "memory")

    f32x8 acc[2][4];
    for (int i = 0; i < 2; ++i)
        for (int j = 0; j < 4; ++j)
            acc[i][j] = (f32x8){0.f,0.f,0.f,0.f,0.f,0.f,0.f,0.f};

#if HAVE_TDM
    if (wave == 0) {   // prologue: chunk 0 -> buffer 0
        tdm_load_tile(0u * tile_bytes, &A [(size_t)m0 * HID], 128, HID);
        tdm_load_tile(1u * tile_bytes, &Bt[(size_t)n0 * HID], 128, HID);
    }
    __builtin_amdgcn_s_wait_tensorcnt(0);
    TDM_LDS_FENCE();
    __syncthreads();
#endif

    for (int it = 0; it < HID / KC; ++it) {
        const int cur = it & 1;
#if HAVE_TDM
        if (wave == 0 && it + 1 < HID / KC) {   // DMA next chunk into back buffer
            const int nxt = cur ^ 1;
            const int kc  = (it + 1) * KC;
            tdm_load_tile((unsigned)(nxt * 2 + 0) * tile_bytes, &A [(size_t)m0 * HID + kc], 128, HID);
            tdm_load_tile((unsigned)(nxt * 2 + 1) * tile_bytes, &Bt[(size_t)n0 * HID + kc], 128, HID);
        }
#else
        {   // synchronous fallback: cooperative 16KB stage into pool[cur]
            const int kc = it * KC;
            for (int i = 0; i < 2; ++i) {
                const int idx = tid + i * 256;       // 0..511
                const int row = idx >> 2;            // 0..127
                const int kg  = (idx & 3) * 8;       // 0,8,16,24
                *(uint4*)&pool[cur][0][row][kg] = *(const uint4*)&A [(size_t)(m0 + row) * HID + kc + kg];
                *(uint4*)&pool[cur][1][row][kg] = *(const uint4*)&Bt[(size_t)(n0 + row) * HID + kc + kg];
            }
            __syncthreads();
        }
#endif
        // ---- compute chunk `it` from pool[cur] ----
        // A frags: lane<16 -> K 0..7 | 16..23 ; lane>=16 -> K 8..15 | 24..31
        bf16x16 a[2], b[4];
        const int ka = hh * 8;
        for (int i = 0; i < 2; ++i) {
            const int row = wm * 32 + i * 16 + l15;
            a[i] = frag_cat(*(const bf16x8*)&pool[cur][0][row][ka],
                            *(const bf16x8*)&pool[cur][0][row][ka + 16]);
        }
        // B frags: lane<16 -> K 0..7 | 16..23 ; lane>=16 -> K 8..15 | 24..31
        const int kb = hh * 8;
        for (int j = 0; j < 4; ++j) {
            const int col = wn * 64 + j * 16 + l15;
            b[j] = frag_cat(*(const bf16x8*)&pool[cur][1][col][kb],
                            *(const bf16x8*)&pool[cur][1][col][kb + 16]);
        }
        for (int i = 0; i < 2; ++i)
            for (int j = 0; j < 4; ++j)
                acc[i][j] = __builtin_amdgcn_wmma_f32_16x16x32_bf16(
                    false, a[i], false, b[j], (short)0, acc[i][j], false, false);

#if HAVE_TDM
        __builtin_amdgcn_s_wait_tensorcnt(0);   // next chunk landed (wave0); no-op others
        TDM_LDS_FENCE();
        __syncthreads();
#else
        __syncthreads();
#endif
    }
#undef TDM_LDS_FENCE

    // epilogue: C layout — lanes 0-15: M=r, N=lane; lanes 16-31: M=8+r, N=lane-16
    for (int i = 0; i < 2; ++i) {
        for (int j = 0; j < 4; ++j) {
            const int col = n0 + wn * 64 + j * 16 + l15;
            if (col < VOCAB) {
                const float bb = bout[col];
                for (int r = 0; r < 8; ++r) {
                    const int row = m0 + wm * 32 + i * 16 + hh * 8 + r;
                    __builtin_nontemporal_store(acc[i][j][r] + bb,
                                                &out[(size_t)row * VOCAB + col]);
                }
            }
        }
    }
}

// =====================================================================
// launcher
// =====================================================================
extern "C" void kernel_launch(void* const* d_in, const int* in_sizes, int n_in,
                              void* d_out, int out_size, void* d_ws, size_t ws_size,
                              hipStream_t stream) {
    const float* x    = (const float*)d_in[0];
    const float* Wg   = (const float*)d_in[1];
    const float* bg   = (const float*)d_in[2];
    const float* We   = (const float*)d_in[3];
    const float* be   = (const float*)d_in[4];
    const float* Wo   = (const float*)d_in[5];
    const float* bo   = (const float*)d_in[6];
    const float* Wout = (const float*)d_in[7];
    const float* bout = (const float*)d_in[8];
    float* out = (float*)d_out;

    char* ws = (char*)d_ws;
    float*  phasor = (float*)(ws);                       // 2048*64*4   = 524288
    float*  mixed  = (float*)(ws + 524288);              // 2048*32*4   = 262144
    float*  ctx    = (float*)(ws + 786432);              // 2048*1024*4 = 8388608
    float*  gains  = (float*)(ws + 9175040);             // 1024*4
    __bf16* Abf    = (__bf16*)(ws + 9179136);            // 2048*1024*2 = 4194304
    __bf16* Bt     = (__bf16*)(ws + 13373440);           // 50304*1024*2 = 103022592

    k_mean_phasor <<<BATCH, 256, 0, stream>>>(x, phasor);
    k_gate_mix    <<<BATCH / 4, 256, 0, stream>>>(x, phasor, Wg, bg, We, be, mixed);
    k_ctx         <<<BATCH, 256, 0, stream>>>(mixed, Wo, bo, ctx);
    k_spike       <<<1, 256, 0, stream>>>(ctx, gains);
    k_attend_bf16 <<<(BATCH * HID) / 256, 256, 0, stream>>>(ctx, gains, Abf);
    k_conv_wout   <<<dim3(NPAD / 64, HID / 64), 256, 0, stream>>>(Wout, Bt);
    k_gemm_wmma   <<<dim3(NPAD / 128, BATCH / 128), 256, 0, stream>>>(Abf, Bt, bout, out);
}
